// CausalSelfAttention_63093069578428
// MI455X (gfx1250) — matrix operand order repaired and checked by reading, not compiled, not run on hardware
//
#include <hip/hip_runtime.h>
#include <math.h>

#define N_EMBD 2048
#define N_HEADS 16
#define N_KVH 4
#define HEAD_DIM 128
#define KVC (N_KVH * HEAD_DIM)   // 512
#define TSEQ 2048
#define BATCH 4
#define MROWS (BATCH * TSEQ)     // 8192

typedef __attribute__((ext_vector_type(16))) __bf16 v16bf;
typedef __attribute__((ext_vector_type(8)))  float  v8f;
typedef __attribute__((ext_vector_type(4)))  unsigned int u32x4;

union FragBF { v16bf bf; u32x4 u[2]; };

#if __has_builtin(__builtin_amdgcn_global_load_async_to_lds_b128) && \
    __has_builtin(__builtin_amdgcn_s_wait_asynccnt)
#define USE_ASYNC_LDS 1
#else
#define USE_ASYNC_LDS 0
#endif

// exact parameter types for the async-to-LDS builtin (from hipcc diagnostic):
// (int4vec AS1*, int4vec AS3*, Imm int, Imm int)
typedef int async_i4 __attribute__((vector_size(4 * sizeof(int))));
typedef __attribute__((address_space(1))) async_i4* as1_i4p;
typedef __attribute__((address_space(3))) async_i4* as3_i4p;

__device__ __forceinline__ unsigned short f2bf(float f) {
  unsigned int u = __float_as_uint(f);
  u += 0x7FFFu + ((u >> 16) & 1u);    // round-to-nearest-even
  return (unsigned short)(u >> 16);
}

__device__ __forceinline__ v8f zero8() {
  v8f z;
  #pragma unroll
  for (int i = 0; i < 8; ++i) z[i] = 0.0f;
  return z;
}

// ---------------------------------------------------------------------------
// fp32 -> bf16 elementwise
// ---------------------------------------------------------------------------
__global__ void cvt_f32_bf16(const float* __restrict__ in,
                             unsigned short* __restrict__ out, int n) {
  int i = blockIdx.x * blockDim.x + threadIdx.x;
  int stride = gridDim.x * blockDim.x;
  for (; i < n; i += stride) out[i] = f2bf(in[i]);
}

// fp32 [K,N] -> bf16 transposed [N,K]  (one-time weight prep)
__global__ void cvt_f32_bf16_T(const float* __restrict__ in,
                               unsigned short* __restrict__ out, int K, int N) {
  int i = blockIdx.x * blockDim.x + threadIdx.x;
  int stride = gridDim.x * blockDim.x;
  int total = K * N;
  for (; i < total; i += stride) {
    int k = i / N, n = i - k * N;
    out[(size_t)n * K + k] = f2bf(in[i]);
  }
}

// ---------------------------------------------------------------------------
// bf16 GEMM, f32 accumulate: C[M,N] = A[M,K] * Bt[N,K]^T
// block: 256 threads (8 waves), tile 128(M) x 128(N), K-step 32
// wave w: (wm = w>>1)*32 rows, (wn = w&1)*64 cols -> 2x4 WMMA subtiles
// A double-buffered in LDS via global_load_async_to_lds_b128 (ASYNCcnt):
// next K-step's tile is issued before computing the current one. B fragments
// direct from global (pre-transposed weights live in L2: 2xB128 per lane).
// MODE: 0 = bf16 row-major, 1 = f32 row-major, 2 = bf16 V-transposed [b][c][t]
// ---------------------------------------------------------------------------
template <int MODE>
__global__ void __launch_bounds__(256)
gemm_bf16_wmma(const unsigned short* __restrict__ A,
               const unsigned short* __restrict__ Bt,
               void* __restrict__ C, int M, int N, int K) {
  __shared__ unsigned short sA[2][128][32];   // ping-pong [buf][m][k]

  const int t    = threadIdx.x;
  const int lane = t & 31;
  const int w    = t >> 5;
  const int wm   = w >> 1, wn = w & 1;
  const int mBase = blockIdx.y * 128;
  const int nBase = blockIdx.x * 128;
  const int hlo   = (lane < 16);
  const int lrow  = lane & 15;

  const int aRowId = t >> 1;
  const int aKh    = (t & 1) * 16;
  const unsigned short* aRow = A + (size_t)(mBase + aRowId) * K + aKh;

  auto stage = [&](int k0, int buf) {
    const unsigned short* gp = aRow + k0;
    unsigned short* sp = &sA[buf][aRowId][aKh];
#if USE_ASYNC_LDS
    __builtin_amdgcn_global_load_async_to_lds_b128(
        (as1_i4p)(void*)gp, (as3_i4p)(void*)sp, 0, 0);
    __builtin_amdgcn_global_load_async_to_lds_b128(
        (as1_i4p)(void*)(gp + 8), (as3_i4p)(void*)(sp + 8), 0, 0);
#else
    const u32x4* g4 = (const u32x4*)gp;
    u32x4* s4 = (u32x4*)sp;
    s4[0] = g4[0];
    s4[1] = g4[1];
#endif
  };
  auto wait_async = [&]() {
#if USE_ASYNC_LDS
    __builtin_amdgcn_s_wait_asynccnt(0);
#endif
  };

  v8f acc[2][4];
  #pragma unroll
  for (int a = 0; a < 2; ++a)
    #pragma unroll
    for (int b = 0; b < 4; ++b) acc[a][b] = zero8();

  stage(0, 0);
  wait_async();
  __syncthreads();

  int cur = 0;
  for (int k0 = 0; k0 < K; k0 += 32) {
    const bool more = (k0 + 32 < K);
    if (more) stage(k0 + 32, cur ^ 1);  // overlap with this step's compute

    FragBF af[2];
    #pragma unroll
    for (int mi = 0; mi < 2; ++mi) {
      int r = wm * 32 + mi * 16 + lrow;
      const u32x4* p = (const u32x4*)(&sA[cur][r][0]);
      // A 16-bit layout: lanes 0-15 K={0..7,16..23}; lanes 16-31 K={8..15,24..31}
      af[mi].u[0] = p[hlo ? 0 : 1];
      af[mi].u[1] = p[hlo ? 2 : 3];
    }
    #pragma unroll
    for (int ni = 0; ni < 4; ++ni) {
      FragBF bfr;
      const unsigned short* brow =
          Bt + (size_t)(nBase + wn * 64 + ni * 16 + lrow) * K + k0;
      const u32x4* p = (const u32x4*)brow;
      // B 16-bit layout: lanes 0-15 K={0..7,16..23}; lanes 16-31 K={8..15,24..31}
      bfr.u[0] = p[hlo ? 0 : 1];
      bfr.u[1] = p[hlo ? 2 : 3];
      #pragma unroll
      for (int mi = 0; mi < 2; ++mi)
        acc[mi][ni] = __builtin_amdgcn_wmma_f32_16x16x32_bf16(
            false, af[mi].bf, false, bfr.bf, (short)0, acc[mi][ni],
            false, false);
    }

    if (more) wait_async();  // own async writes to next buffer done
    __syncthreads();         // everyone's writes visible; cur reads finished
    cur ^= 1;
  }

  // epilogue: C/D layout -> VGPR v holds row v (lanes 0-15) / v+8 (lanes 16-31)
  #pragma unroll
  for (int mi = 0; mi < 2; ++mi)
    #pragma unroll
    for (int ni = 0; ni < 4; ++ni)
      #pragma unroll
      for (int v = 0; v < 8; ++v) {
        int row = mBase + wm * 32 + mi * 16 + v + (hlo ? 0 : 8);
        int col = nBase + wn * 64 + ni * 16 + lrow;
        float val = acc[mi][ni][v];
        if (MODE == 1) {
          ((float*)C)[(size_t)row * N + col] = val;
        } else if (MODE == 0) {
          ((unsigned short*)C)[(size_t)row * N + col] = f2bf(val);
        } else {  // V transposed: [b][c][t], c in [0,512), t in [0,2048)
          int bb = row >> 11, tt = row & 2047;
          ((unsigned short*)C)[(size_t)bb * KVC * TSEQ + (size_t)col * TSEQ + tt] =
              f2bf(val);
        }
      }
}

// ---------------------------------------------------------------------------
// Flash attention with ALiBi + causal mask, GQA (4 KV heads).
// block: 128 threads = 4 independent waves; 16 q rows per wave.
// K fragments and pre-transposed V fragments load directly from global.
// Only LDS use: per-wave P re-layout (C-layout -> A-fragment).
// ALiBi decomposed: slope*(j-i) = slope*j - slope*i with slope*i hoisted.
// ---------------------------------------------------------------------------
__global__ void __launch_bounds__(128)
attn_flash_wmma(const unsigned short* __restrict__ Q,   // [B*T, 2048]
                const unsigned short* __restrict__ Km,  // [B*T, 512]
                const unsigned short* __restrict__ Vt,  // [B][512][T]
                unsigned short* __restrict__ Y) {       // [B*T, 2048]
  __shared__ unsigned short sP[4][16][32];   // per-wave P tile

  const int t    = threadIdx.x;
  const int lane = t & 31;
  const int w    = t >> 5;
  const int hlo  = (lane < 16);
  const int lrow = lane & 15;

  const int bh  = blockIdx.x;        // B*H = 64
  const int b   = bh >> 4;
  const int h   = bh & 15;
  const int kvh = h >> 2;
  const int qBase = blockIdx.y * 64 + w * 16;

  const float slope    = exp2f(-0.5f * (float)(kvh + 1));  // repeat_interleave slopes
  const float sm_scale = 0.08838834764831845f;             // 1/sqrt(128)

  // preload Q fragments (A layout), reused across all key blocks
  FragBF qf[4];
  {
    int qi = qBase + lrow;
    const unsigned short* qrow =
        Q + (size_t)(b * TSEQ + qi) * N_EMBD + h * HEAD_DIM;
    #pragma unroll
    for (int c = 0; c < 4; ++c) {
      const u32x4* p = (const u32x4*)(qrow + c * 32);
      qf[c].u[0] = p[hlo ? 0 : 1];
      qf[c].u[1] = p[hlo ? 2 : 3];
    }
  }

  v8f o[8];
  #pragma unroll
  for (int i = 0; i < 8; ++i) o[i] = zero8();
  float m[8], l[8];
  int   iv[8];
  float fi[8];
  #pragma unroll
  for (int v = 0; v < 8; ++v) {
    m[v] = -1e30f;
    l[v] = 0.0f;
    iv[v] = qBase + v + (hlo ? 0 : 8);       // this lane's row index per VGPR
    fi[v] = slope * (float)iv[v];            // hoisted slope*i
  }

  const unsigned short* vtBase = Vt + (size_t)b * KVC * TSEQ;

  for (int j0 = 0; j0 <= qBase + 15; j0 += 32) {
    // S = Q K^T  (two 16x16 n-tiles over 32 keys; K^T frags direct from global)
    v8f s[2];
    #pragma unroll
    for (int nt = 0; nt < 2; ++nt) {
      v8f sacc = zero8();
      int kj = j0 + nt * 16 + lrow;
      const unsigned short* krow =
          Km + (size_t)(b * TSEQ + kj) * KVC + kvh * HEAD_DIM;
      #pragma unroll
      for (int c = 0; c < 4; ++c) {
        FragBF kf;
        const u32x4* p = (const u32x4*)(krow + c * 32);
        kf.u[0] = p[hlo ? 0 : 1];
        kf.u[1] = p[hlo ? 2 : 3];
        sacc = __builtin_amdgcn_wmma_f32_16x16x32_bf16(
            false, qf[c].bf, false, kf.bf, (short)0, sacc, false, false);
      }
      s[nt] = sacc;
    }

    // scale + ALiBi + causal mask (slope*j computed once per n-tile)
    #pragma unroll
    for (int nt = 0; nt < 2; ++nt) {
      int jj = j0 + nt * 16 + lrow;          // this lane's key column
      float fj = slope * (float)jj;
      #pragma unroll
      for (int v = 0; v < 8; ++v) {
        float val = fmaf(s[nt][v], sm_scale, fj - fi[v]);
        s[nt][v] = (jj <= iv[v]) ? val : -1e30f;
      }
    }

    // online softmax; row v lives across 16 lanes of each half-wave
    #pragma unroll
    for (int v = 0; v < 8; ++v) {
      float mx = fmaxf(s[0][v], s[1][v]);
      #pragma unroll
      for (int off = 8; off >= 1; off >>= 1)
        mx = fmaxf(mx, __shfl_xor(mx, off, 16));
      float mnew = fmaxf(m[v], mx);
      float corr = __expf(m[v] - mnew);
      float p0 = __expf(s[0][v] - mnew);
      float p1 = __expf(s[1][v] - mnew);
      s[0][v] = p0; s[1][v] = p1;
      float rs = p0 + p1;
      #pragma unroll
      for (int off = 8; off >= 1; off >>= 1) rs += __shfl_xor(rs, off, 16);
      l[v] = l[v] * corr + rs;
      m[v] = mnew;
      #pragma unroll
      for (int d = 0; d < 8; ++d) o[d][v] *= corr;
    }

    // P (C layout) -> LDS bf16 -> reload as A fragment (same-wave, no barrier)
    #pragma unroll
    for (int nt = 0; nt < 2; ++nt)
      #pragma unroll
      for (int v = 0; v < 8; ++v)
        sP[w][v + (hlo ? 0 : 8)][nt * 16 + lrow] = f2bf(s[nt][v]);
    asm volatile("s_wait_dscnt 0" ::: "memory");  // same-wave LDS RAW

    FragBF pf;
    {
      const u32x4* p = (const u32x4*)(&sP[w][lrow][0]);
      pf.u[0] = p[hlo ? 0 : 1];
      pf.u[1] = p[hlo ? 2 : 3];
    }

    // O += P @ V  (8 d-tiles; V^T fragments contiguous from global)
    #pragma unroll
    for (int nt = 0; nt < 8; ++nt) {
      FragBF vf;
      const unsigned short* vcol =
          vtBase + (size_t)(kvh * HEAD_DIM + nt * 16 + lrow) * TSEQ + j0;
      const u32x4* p = (const u32x4*)vcol;
      vf.u[0] = p[hlo ? 0 : 1];
      vf.u[1] = p[hlo ? 2 : 3];
      o[nt] = __builtin_amdgcn_wmma_f32_16x16x32_bf16(
          false, pf.bf, false, vf.bf, (short)0, o[nt], false, false);
    }
  }

  // normalize and store y (bf16) at [b*T+i, h*128 + d]
  #pragma unroll
  for (int v = 0; v < 8; ++v) {
    float inv = 1.0f / l[v];
    int i = iv[v];
    unsigned short* yrow =
        Y + (size_t)(b * TSEQ + i) * N_EMBD + h * HEAD_DIM;
    #pragma unroll
    for (int nt = 0; nt < 8; ++nt)
      yrow[nt * 16 + lrow] = f2bf(o[nt][v] * inv);
  }
}

// ---------------------------------------------------------------------------
extern "C" void kernel_launch(void* const* d_in, const int* in_sizes, int n_in,
                              void* d_out, int out_size, void* d_ws,
                              size_t ws_size, hipStream_t stream) {
  (void)in_sizes; (void)n_in; (void)out_size; (void)ws_size;
  const float* x  = (const float*)d_in[0];
  const float* Wq = (const float*)d_in[1];
  const float* Wk = (const float*)d_in[2];
  const float* Wv = (const float*)d_in[3];
  const float* Wo = (const float*)d_in[4];
  float* out = (float*)d_out;

  char* ws = (char*)d_ws;
  size_t off = 0;
  auto alloc = [&](size_t bytes) -> char* {
    char* p = ws + off;
    off += (bytes + 255) & ~(size_t)255;
    return p;
  };
  unsigned short* xb  = (unsigned short*)alloc((size_t)MROWS * N_EMBD * 2);
  unsigned short* wqt = (unsigned short*)alloc((size_t)N_EMBD * N_EMBD * 2);
  unsigned short* wkt = (unsigned short*)alloc((size_t)N_EMBD * KVC * 2);
  unsigned short* wvt = (unsigned short*)alloc((size_t)N_EMBD * KVC * 2);
  unsigned short* wot = (unsigned short*)alloc((size_t)N_EMBD * N_EMBD * 2);
  unsigned short* qb  = (unsigned short*)alloc((size_t)MROWS * N_EMBD * 2);
  unsigned short* kb  = (unsigned short*)alloc((size_t)MROWS * KVC * 2);
  unsigned short* vtb = (unsigned short*)alloc((size_t)MROWS * KVC * 2);
  unsigned short* yb  = (unsigned short*)alloc((size_t)MROWS * N_EMBD * 2);

  cvt_f32_bf16<<<1024, 256, 0, stream>>>(x, xb, MROWS * N_EMBD);
  cvt_f32_bf16_T<<<512, 256, 0, stream>>>(Wq, wqt, N_EMBD, N_EMBD);
  cvt_f32_bf16_T<<<256, 256, 0, stream>>>(Wk, wkt, N_EMBD, KVC);
  cvt_f32_bf16_T<<<256, 256, 0, stream>>>(Wv, wvt, N_EMBD, KVC);
  cvt_f32_bf16_T<<<512, 256, 0, stream>>>(Wo, wot, N_EMBD, N_EMBD);

  dim3 gq(N_EMBD / 128, MROWS / 128);
  dim3 gkv(KVC / 128, MROWS / 128);
  gemm_bf16_wmma<0><<<gq, 256, 0, stream>>>(xb, wqt, qb, MROWS, N_EMBD, N_EMBD);
  gemm_bf16_wmma<0><<<gkv, 256, 0, stream>>>(xb, wkt, kb, MROWS, KVC, N_EMBD);
  gemm_bf16_wmma<2><<<gkv, 256, 0, stream>>>(xb, wvt, vtb, MROWS, KVC, N_EMBD);

  dim3 ga(BATCH * N_HEADS, TSEQ / 64);
  attn_flash_wmma<<<ga, 128, 0, stream>>>(qb, kb, vtb, yb);

  gemm_bf16_wmma<1><<<gq, 256, 0, stream>>>(yb, wot, out, MROWS, N_EMBD, N_EMBD);
}
